// PolicyMLP_37769942401530
// MI455X (gfx1250) — compile-verified, matrix operand order checked by reading, not run on hardware
//
#include <hip/hip_runtime.h>
#include <math.h>

#define S     2048
#define NBITS 256
#define MID   512
#define RPB   32      // rows (sims) per block
#define LDST  516     // padded LDS row stride in floats (8*516 mod 64 = 32 -> conflict-free)

typedef __attribute__((ext_vector_type(16))) __bf16 v16bf;
typedef __attribute__((ext_vector_type(8)))  __bf16 v8bf;
typedef __attribute__((ext_vector_type(8)))  float  v8f;

__device__ __forceinline__ unsigned short f2bf(float f) {
  union { float f; unsigned u; } v; v.f = f;
  unsigned r = v.u + 0x7FFFu + ((v.u >> 16) & 1u);  // round-to-nearest-even
  return (unsigned short)(r >> 16);
}

// Load a 16x32 bf16 A/B fragment slice for this lane: two contiguous 16-byte loads.
// Layout per CDNA5 ISA: lanes<16 hold K=kb..kb+7 and kb+16..kb+23 with kb=0,
// lanes>=16 the same with kb=8. Row (A: M, B: N) = lane&15.
__device__ __forceinline__ v16bf load_frag(const unsigned short* p) {
  v8bf lo = *(const v8bf*)p;
  v8bf hi = *(const v8bf*)(p + 16);
  v16bf r;
#pragma unroll
  for (int i = 0; i < 8; ++i) { r[i] = lo[i]; r[i + 8] = hi[i]; }
  return r;
}

// One-time setup: probs (f32 + bf16 mirror), w1 transposed->bf16, winner slab = -1.
__global__ void k_init(const float* __restrict__ xs, float* __restrict__ probs,
                       unsigned short* __restrict__ probs_bf,
                       const float* __restrict__ w1, unsigned short* __restrict__ w1T,
                       int* __restrict__ winner) {
  int i = blockIdx.x * blockDim.x + threadIdx.x;
  if (i < S * NBITS) { float v = xs[i]; probs[i] = v; probs_bf[i] = f2bf(v); }
  if (i < MID * NBITS) {
    int n = i / NBITS, k = i % NBITS;
    w1T[n * NBITS + k] = f2bf(w1[k * MID + n]);   // w1T[n][k] = w1[k][n]
  }
  if (i < NBITS * NBITS) winner[i] = -1;
}

// Per-step: h = LN(GELU(probs@w1 + b1)); ps = sigmoid(<h_row, w2[:,id]> + b2[id]);
// record last-winner per column for the broadcast scatter.
__global__ void k_step_fwd(const unsigned short* __restrict__ probs_bf,
                           const unsigned short* __restrict__ w1T,
                           const float* __restrict__ b1,
                           const float* __restrict__ gamma,
                           const float* __restrict__ beta,
                           const float* __restrict__ w2,
                           const float* __restrict__ b2,
                           const int* __restrict__ ids,   // [S][NBITS]
                           int step,
                           float* __restrict__ ps,        // [S]
                           int* __restrict__ winner)      // [NBITS], this step's slab
{
  extern __shared__ float h[];                   // [RPB][LDST]
  const int tid  = threadIdx.x;
  const int wave = tid >> 5;
  const int lane = tid & 31;
  const int row0 = blockIdx.x * RPB;

  // ---- GEMM1 via WMMA: wave -> 16 rows x 8 n-tiles of 16 ----
  const int mt  = wave >> 2;                     // 0..1 (which 16-row half)
  const int nt0 = (wave & 3) * 8;                // n-tile base (8 tiles each)
  const int m   = lane & 15;
  const int kb  = (lane >> 4) * 8;
  const unsigned short* arow =
      probs_bf + (size_t)(row0 + mt * 16 + m) * NBITS + kb;
#pragma unroll
  for (int nt = 0; nt < 8; ++nt) {
    const int n0 = (nt0 + nt) * 16;
    const unsigned short* brow = w1T + (size_t)(n0 + m) * NBITS + kb;
    v8f acc = {};
#pragma unroll
    for (int k = 0; k < NBITS; k += 32) {
      v16bf a = load_frag(arow + k);
      v16bf b = load_frag(brow + k);
      acc = __builtin_amdgcn_wmma_f32_16x16x32_bf16(
          false, a, false, b, (short)0, acc, false, false);
    }
    // C/D layout: VGPR i -> M = i + 8*(lane>=16), N = lane&15
    const int mrow = mt * 16 + (lane >> 4) * 8;
#pragma unroll
    for (int i = 0; i < 8; ++i)
      h[(mrow + i) * LDST + n0 + m] = acc[i];
  }
  __syncthreads();

  // ---- GELU(exact erf) + LayerNorm + gathered layer-2 dot + sigmoid ----
  for (int rr = 0; rr < 4; ++rr) {
    const int r = wave * 4 + rr;                 // 8 waves * 4 rows = 32 rows
    float sum = 0.f, sumsq = 0.f;
#pragma unroll
    for (int i = 0; i < 16; ++i) {
      const int c = i * 32 + lane;               // lane-adjacent: bank conflict free
      float x = h[r * LDST + c] + b1[c];
      float g = 0.5f * x * (1.f + erff(x * 0.70710678118654752f));
      h[r * LDST + c] = g;
      sum += g; sumsq += g * g;
    }
#pragma unroll
    for (int off = 16; off > 0; off >>= 1) {
      sum   += __shfl_xor(sum, off, 32);
      sumsq += __shfl_xor(sumsq, off, 32);
    }
    const float mean = sum * (1.f / MID);
    const float var  = sumsq * (1.f / MID) - mean * mean;
    const float rstd = rsqrtf(var + 1e-5f);

    const int sim = row0 + r;
    const int id  = ids[(size_t)sim * NBITS + step];
    float dot = 0.f;
#pragma unroll
    for (int i = 0; i < 16; ++i) {
      const int c  = i * 32 + lane;
      float hn = (h[r * LDST + c] - mean) * rstd * gamma[c] + beta[c];
      dot += hn * w2[(size_t)c * NBITS + id];
    }
#pragma unroll
    for (int off = 16; off > 0; off >>= 1) dot += __shfl_xor(dot, off, 32);
    if (lane == 0) {
      float p = 1.f / (1.f + expf(-(dot + b2[id])));
      ps[sim] = p;
      atomicMax(&winner[id], sim);               // torch last-write-wins == max sim
    }
  }
}

// Broadcast scatter: probs[:, c] = ps[winner[c]] for every written column, f32 + bf16.
__global__ void k_scatter(const int* __restrict__ winner,
                          const float* __restrict__ ps,
                          float* __restrict__ probs,
                          unsigned short* __restrict__ probs_bf) {
  const int c = threadIdx.x;                     // 256 threads = 256 columns
  const int row0 = blockIdx.x * RPB;
  const int j = winner[c];
  if (j < 0) return;
  const float v = ps[j];
  const unsigned short vb = f2bf(v);
  for (int r = 0; r < RPB; ++r) {
    probs[(size_t)(row0 + r) * NBITS + c]    = v;
    probs_bf[(size_t)(row0 + r) * NBITS + c] = vb;
  }
}

extern "C" void kernel_launch(void* const* d_in, const int* in_sizes, int n_in,
                              void* d_out, int out_size, void* d_ws, size_t ws_size,
                              hipStream_t stream) {
  const float* xs    = (const float*)d_in[0];
  const int*   ids   = (const int*)  d_in[1];
  const float* w1    = (const float*)d_in[2];
  const float* b1    = (const float*)d_in[3];
  const float* gamma = (const float*)d_in[4];
  const float* beta  = (const float*)d_in[5];
  const float* w2    = (const float*)d_in[6];
  const float* b2    = (const float*)d_in[7];
  float* probs = (float*)d_out;                  // d_out doubles as the live probs

  char* ws = (char*)d_ws;
  unsigned short* probs_bf = (unsigned short*)ws; ws += (size_t)S * NBITS * 2;
  unsigned short* w1T      = (unsigned short*)ws; ws += (size_t)MID * NBITS * 2;
  float*          ps       = (float*)ws;          ws += (size_t)S * 4;
  int*            winner   = (int*)ws;            // [NBITS steps][NBITS cols]

  k_init<<<dim3((S * NBITS + 255) / 256), dim3(256), 0, stream>>>(
      xs, probs, probs_bf, w1, w1T, winner);

  const size_t ldsB = (size_t)RPB * LDST * sizeof(float);  // ~64.5 KB of 320 KB/WGP
  for (int t = 0; t < NBITS; ++t) {
    k_step_fwd<<<dim3(S / RPB), dim3(256), ldsB, stream>>>(
        probs_bf, w1T, b1, gamma, beta, w2, b2, ids, t, ps, winner + t * NBITS);
    k_scatter<<<dim3(S / RPB), dim3(256), 0, stream>>>(
        winner + t * NBITS, ps, probs, probs_bf);
  }
}
